// CrossAttention_1726576857384
// MI455X (gfx1250) — matrix operand order repaired and checked by reading, not compiled, not run on hardware
//
#include <hip/hip_runtime.h>
#include <hip/hip_bf16.h>

// ---------------- problem constants ----------------
#define BB 4
#define NN 1024
#define LL 1024
#define CC 1024
#define HH 16
#define DD 64
#define SCALE_F 0.125f
#define NEG_F (-65504.0f)

typedef __attribute__((ext_vector_type(16))) __bf16 v16bf;
typedef __attribute__((ext_vector_type(8)))  float  v8f;
typedef __attribute__((ext_vector_type(4)))  unsigned int v4u;
typedef __attribute__((ext_vector_type(8)))  int v8i;
typedef __attribute__((ext_vector_type(4)))  int v4i;

#if defined(__has_builtin)
#if __has_builtin(__builtin_amdgcn_tensor_load_to_lds) && __has_builtin(__builtin_amdgcn_s_wait_tensorcnt)
#define USE_TDM 1
#endif
#endif

union Frag { unsigned int u[8]; v16bf v; };

__device__ __forceinline__ unsigned short f2bf(float f) {
  unsigned int u = __float_as_uint(f);
  u += 0x7FFFu + ((u >> 16) & 1u);            // round-to-nearest-even
  return (unsigned short)(u >> 16);
}
__device__ __forceinline__ float bf2f(unsigned short s) {
  return __uint_as_float(((unsigned int)s) << 16);
}

// A-fragment (16x32 bf16): lane m = lane%16; VGPR j holds K pair starting at:
//   j<4 : K = 2j + 8*(lane/16) ; j>=4 : K = 2j+8 + 8*(lane/16)
__device__ __forceinline__ int kbA(int j, int g) { return ((j < 4) ? 2 * j : 2 * j + 8) + 8 * g; }
// B-fragment (32x16 bf16): lane n = lane%16; VGPR j holds K pair starting at:
//   j<4 : K = 2j + 8*(lane/16) ; j>=4 : K = 2j+8 + 8*(lane/16)
__device__ __forceinline__ int kbB(int j, int g) { return ((j < 4) ? 2 * j : 2 * j + 8) + 8 * g; }

#ifdef USE_TDM
// TDM: async-copy a 64(rows) x 64(bf16) tile, row stride CC elements, into LDS.
// D# per CDNA5 ISA: group0 = {count/lds_addr/global_addr/type},
// group1 = {data_size, tensor/tile dims, dim0 stride}. 2D tensor -> remaining
// descriptor groups zero. This toolchain exposes the 6-arg builtin:
//   (v4u g0, v8i g1, v4i g2, v4i g3, v8i g4, i32 cpol)
__device__ __forceinline__ void tdm_load_tile64x64(const unsigned short* gptr,
                                                   unsigned int lds_off) {
  unsigned long long ga = (unsigned long long)(size_t)gptr;
  v4u g0;
  g0[0] = 1u;                                        // count=1 (valid descriptor)
  g0[1] = lds_off;                                   // lds_addr (bytes)
  g0[2] = (unsigned int)ga;                          // global_addr[31:0]
  g0[3] = (unsigned int)((ga >> 32) & 0x1FFFFFFu)    // global_addr[56:32]
          | (2u << 30);                              // type=2 ("image")
  v8i g1;
  g1[0] = (int)(1u << 16);                           // data_size=1 (2 bytes)
  g1[1] = (int)(64u << 16);                          // tensor_dim0[15:0]=64
  g1[2] = (int)(64u << 16);                          // tensor_dim1[15:0]=64
  g1[3] = (int)(64u << 16);                          // tile_dim0=64
  g1[4] = (int)(64u);                                // tile_dim1=64, tile_dim2=0
  g1[5] = (int)(1024u);                              // tensor_dim0_stride[31:0]=CC
  g1[6] = 0;                                         // strides hi / dim1_stride
  g1[7] = 0;
  v4i gz4; gz4[0] = 0; gz4[1] = 0; gz4[2] = 0; gz4[3] = 0;
  v8i gz8;
#pragma unroll
  for (int j = 0; j < 8; ++j) gz8[j] = 0;
  __builtin_amdgcn_tensor_load_to_lds(g0, g1, gz4, gz4, gz8, 0);
}
#endif

// ---------------- f32 -> bf16 convert (vectorized) ----------------
__global__ void cvt_bf16_kernel(const float* __restrict__ in,
                                unsigned short* __restrict__ out, int n) {
  int i = (blockIdx.x * blockDim.x + threadIdx.x) * 4;
  if (i + 3 < n) {
    float4 f = *(const float4*)(in + i);
    unsigned int lo = (unsigned int)f2bf(f.x) | ((unsigned int)f2bf(f.y) << 16);
    unsigned int hi = (unsigned int)f2bf(f.z) | ((unsigned int)f2bf(f.w) << 16);
    uint2 p; p.x = lo; p.y = hi;
    *(uint2*)(out + i) = p;
  }
}

// ---------------- generic bf16 GEMM: out = A(MxK) @ W(NoutxK)^T ----------------
// mode 0: store bf16 to outBf[row*Nout + o]                       (Q proj)
// mode 1: KV split store: o<CC -> K row-major; o>=CC -> V transposed (B,H,D,L)
// mode 2: store f32 + bias to outF[row*Nout + o]                  (out proj)
__global__ void gemm_bf16_kernel(const unsigned short* __restrict__ A,
                                 const unsigned short* __restrict__ W,
                                 int K, int Nout, int nTiles, int mode,
                                 unsigned short* __restrict__ outBf,
                                 unsigned short* __restrict__ outK,
                                 unsigned short* __restrict__ outVt,
                                 float* __restrict__ outF,
                                 const float* __restrict__ bias) {
  const int wv   = threadIdx.x >> 5;
  const int lane = threadIdx.x & 31;
  const int g    = lane >> 4;
  const int col  = lane & 15;

  const int tile = blockIdx.x * 4 + wv;
  const int tm = tile / nTiles;
  const int tn = tile % nTiles;

  const unsigned short* Arow = A + (size_t)(tm * 16 + col) * K;   // A row m = col
  const unsigned short* Wrow = W + (size_t)(tn * 16 + col) * K;   // B column n = col

  v8f acc = {};
  for (int kc = 0; kc < K; kc += 32) {
    Frag a, b;
#pragma unroll
    for (int j = 0; j < 8; ++j) {
      a.u[j] = *(const unsigned int*)(Arow + kc + kbA(j, g));
      b.u[j] = *(const unsigned int*)(Wrow + kc + kbB(j, g));
    }
    acc = __builtin_amdgcn_wmma_f32_16x16x32_bf16(false, a.v, false, b.v,
                                                  (short)0, acc, false, false);
  }

#pragma unroll
  for (int i = 0; i < 8; ++i) {
    const int m   = i + 8 * g;
    const int row = tm * 16 + m;
    const int o   = tn * 16 + col;
    const float v = acc[i];
    if (mode == 0) {
      outBf[(size_t)row * Nout + o] = f2bf(v);
    } else if (mode == 1) {
      const int b = row >> 10;            // row = b*LL + l
      const int l = row & (LL - 1);
      if (o < CC) {
        outK[((size_t)b * LL + l) * CC + o] = f2bf(v);
      } else {
        const int oh = o - CC;
        const int h  = oh >> 6;
        const int d  = oh & 63;
        outVt[(((size_t)b * HH + h) * DD + d) * LL + l] = f2bf(v);
      }
    } else {
      outF[(size_t)row * Nout + o] = v + bias[o];
    }
  }
}

// ---------------- fused attention: scores + mask + bias + softmax + PV ----------------
// block = (b, h, nt): 16 query rows; 128 threads = 4 waves.
// K tiles are staged into LDS by the Tensor Data Mover (double buffered, one chunk
// ahead), synchronized via TENSORcnt + workgroup barrier.
__global__ void attn_kernel(const unsigned short* __restrict__ qbf,   // (B,N,C) bf16
                            const unsigned short* __restrict__ kbf,   // (B,L,C) bf16
                            const unsigned short* __restrict__ vt,    // (B,H,D,L) bf16
                            const float* __restrict__ bias,           // (B,H,N,L)
                            const unsigned char* __restrict__ mask,   // (B,N,L) bool
                            float* __restrict__ attn_out,             // (B,H,N,L)
                            unsigned short* __restrict__ ctx_bf) {    // (B,N,C) bf16
  __shared__ unsigned short Sld[16 * LL];   // scores / probs, bf16: 32 KB
#ifdef USE_TDM
  __shared__ unsigned short Kst[2][64 * 64]; // double-buffered K tiles: 16 KB
#endif
  __shared__ float red[128];
  __shared__ float rowmax[16];
  __shared__ float rowinv[16];

  const int blk = blockIdx.x;
  const int nt = blk & 63;
  const int h  = (blk >> 6) & 15;
  const int b  = blk >> 10;

  const int wv   = threadIdx.x >> 5;
  const int lane = threadIdx.x & 31;
  const int g    = lane >> 4;
  const int col  = lane & 15;

  // Q A-fragments for this 16-row block: resident in registers for all of L.
  const unsigned short* qrow = qbf + ((size_t)b * NN + nt * 16 + col) * CC + h * DD;
  Frag aLo, aHi;
#pragma unroll
  for (int j = 0; j < 8; ++j) {
    aLo.u[j] = *(const unsigned int*)(qrow + kbA(j, g));
    aHi.u[j] = *(const unsigned int*)(qrow + 32 + kbA(j, g));
  }

  // ---- S = scale*Q K^T + bias, masked; raw to HBM, bf16 to LDS ----
#ifdef USE_TDM
  const unsigned short* kchunk0 = kbf + ((size_t)b * LL + 0) * CC + h * DD;
  if (wv == 0) {
    tdm_load_tile64x64(kchunk0, (unsigned int)(size_t)(void*)&Kst[0][0]);
  }
  for (int it = 0; it < LL / 64; ++it) {
    if (wv == 0) {
      if (it + 1 < LL / 64) {
        const unsigned short* knext =
            kbf + ((size_t)b * LL + (it + 1) * 64) * CC + h * DD;
        tdm_load_tile64x64(knext, (unsigned int)(size_t)(void*)&Kst[(it + 1) & 1][0]);
        __builtin_amdgcn_s_wait_tensorcnt(1);   // tile `it` complete
      } else {
        __builtin_amdgcn_s_wait_tensorcnt(0);
      }
    }
    __syncthreads();
    const int l0 = it * 64 + wv * 16;
    const unsigned short* krow = &Kst[it & 1][(wv * 16 + col) * 64];
    Frag b0, b1;
#pragma unroll
    for (int j = 0; j < 8; ++j) {
      b0.u[j] = *(const unsigned int*)(krow + kbB(j, g));
      b1.u[j] = *(const unsigned int*)(krow + 32 + kbB(j, g));
    }
#else
  for (int it = 0; it < LL / 64; ++it) {
    const int l0 = it * 64 + wv * 16;
    const unsigned short* krow = kbf + ((size_t)b * LL + l0 + col) * CC + h * DD;
    Frag b0, b1;
#pragma unroll
    for (int j = 0; j < 8; ++j) {
      b0.u[j] = *(const unsigned int*)(krow + kbB(j, g));
      b1.u[j] = *(const unsigned int*)(krow + 32 + kbB(j, g));
    }
#endif
    v8f acc = {};
    acc = __builtin_amdgcn_wmma_f32_16x16x32_bf16(false, aLo.v, false, b0.v,
                                                  (short)0, acc, false, false);
    acc = __builtin_amdgcn_wmma_f32_16x16x32_bf16(false, aHi.v, false, b1.v,
                                                  (short)0, acc, false, false);
#pragma unroll
    for (int i = 0; i < 8; ++i) {
      const int m  = i + 8 * g;
      const int nq = nt * 16 + m;
      const int lg = l0 + col;
      float s = acc[i] * SCALE_F;
      if (mask[((size_t)b * NN + nq) * LL + lg]) s = NEG_F;
      s += bias[(((size_t)b * HH + h) * NN + nq) * LL + lg];
      attn_out[(((size_t)b * HH + h) * NN + nq) * LL + lg] = s;
      Sld[m * LL + lg] = f2bf(s);
    }
#ifdef USE_TDM
    __syncthreads();   // tile consumed; its buffer may be overwritten next iter
#endif
  }
  __syncthreads();

  // ---- softmax over L in LDS: 8 threads per row ----
  {
    const int r   = threadIdx.x >> 3;
    const int idx = threadIdx.x & 7;
    const int c0  = idx * (LL / 8);
    float mx = -3.4e38f;
    for (int c = 0; c < LL / 8; ++c) mx = fmaxf(mx, bf2f(Sld[r * LL + c0 + c]));
    red[threadIdx.x] = mx;
    __syncthreads();
    if (idx == 0) {
      float m2 = red[threadIdx.x];
#pragma unroll
      for (int k = 1; k < 8; ++k) m2 = fmaxf(m2, red[threadIdx.x + k]);
      rowmax[r] = m2;
    }
    __syncthreads();
    const float rm = rowmax[r];
    float sum = 0.0f;
    for (int c = 0; c < LL / 8; ++c) {
      const float e = __expf(bf2f(Sld[r * LL + c0 + c]) - rm);
      sum += e;
      Sld[r * LL + c0 + c] = f2bf(e);
    }
    red[threadIdx.x] = sum;
    __syncthreads();
    if (idx == 0) {
      float s2 = red[threadIdx.x];
#pragma unroll
      for (int k = 1; k < 8; ++k) s2 += red[threadIdx.x + k];
      rowinv[r] = 1.0f / s2;
    }
    __syncthreads();
  }

  // ---- ctx = P @ V ; wave wv owns d-range [wv*16, wv*16+16) ----
  // V is (B,H,D,L): per-lane two 16B-contiguous runs of l-pairs; L2-resident, read direct.
  {
    v8f acc = {};
    const unsigned short* vbase = vt + (((size_t)b * HH + h) * DD + wv * 16 + col) * LL;
    for (int kc = 0; kc < LL; kc += 32) {
      Frag a, vb;
#pragma unroll
      for (int j = 0; j < 8; ++j) {
        a.u[j]  = *(const unsigned int*)&Sld[col * LL + kc + kbA(j, g)];
        vb.u[j] = *(const unsigned int*)(vbase + kc + kbB(j, g));
      }
      acc = __builtin_amdgcn_wmma_f32_16x16x32_bf16(false, a.v, false, vb.v,
                                                    (short)0, acc, false, false);
    }
#pragma unroll
    for (int i = 0; i < 8; ++i) {
      const int m = i + 8 * g;
      const float v = acc[i] * rowinv[m];
      ctx_bf[((size_t)b * NN + nt * 16 + m) * CC + h * DD + wv * 16 + col] = f2bf(v);
    }
  }
}

// ---------------- host launch ----------------
extern "C" void kernel_launch(void* const* d_in, const int* in_sizes, int n_in,
                              void* d_out, int out_size, void* d_ws, size_t ws_size,
                              hipStream_t stream) {
  const float* query  = (const float*)d_in[0];
  const float* memory = (const float*)d_in[1];
  const float* bias   = (const float*)d_in[2];
  const unsigned char* mask = (const unsigned char*)d_in[3]; // jax bool_
  const float* q_w    = (const float*)d_in[4];
  const float* kv_w   = (const float*)d_in[5];
  const float* proj_w = (const float*)d_in[6];
  const float* proj_b = (const float*)d_in[7];

  float* out_x    = (float*)d_out;                                 // (B,N,C)
  float* out_attn = (float*)d_out + (size_t)BB * NN * CC;          // (B,H,N,L)

  const size_t MB = 1024u * 1024u;
  char* ws = (char*)d_ws;
  unsigned short* query_bf = (unsigned short*)(ws +  0 * MB);  // 8 MB
  unsigned short* mem_bf   = (unsigned short*)(ws +  8 * MB);  // 8 MB
  unsigned short* qw_bf    = (unsigned short*)(ws + 16 * MB);  // 2 MB
  unsigned short* kvw_bf   = (unsigned short*)(ws + 18 * MB);  // 4 MB
  unsigned short* pw_bf    = (unsigned short*)(ws + 22 * MB);  // 2 MB
  unsigned short* q_bf     = (unsigned short*)(ws + 24 * MB);  // 8 MB
  unsigned short* k_bf     = (unsigned short*)(ws + 32 * MB);  // 8 MB
  unsigned short* vt_bf    = (unsigned short*)(ws + 40 * MB);  // 8 MB (B,H,D,L)
  unsigned short* ctx_bf   = (unsigned short*)(ws + 48 * MB);  // 8 MB
  // total 56 MB of workspace

  // f32 -> bf16 conversions
  cvt_bf16_kernel<<<4096, 256, 0, stream>>>(query,  query_bf, BB * NN * CC);
  cvt_bf16_kernel<<<4096, 256, 0, stream>>>(memory, mem_bf,   BB * LL * CC);
  cvt_bf16_kernel<<<1024, 256, 0, stream>>>(q_w,    qw_bf,    CC * CC);
  cvt_bf16_kernel<<<2048, 256, 0, stream>>>(kv_w,   kvw_bf,   2 * CC * CC);
  cvt_bf16_kernel<<<1024, 256, 0, stream>>>(proj_w, pw_bf,    CC * CC);

  // Q projection: (4096 x 1024) @ (1024 x 1024)^T -> q_bf
  gemm_bf16_kernel<<<(256 * 64) / 4, 128, 0, stream>>>(
      query_bf, qw_bf, CC, CC, 64, 0, q_bf, nullptr, nullptr, nullptr, nullptr);

  // KV projection: (4096 x 1024) @ (2048 x 1024)^T -> k_bf row-major, v transposed
  gemm_bf16_kernel<<<(256 * 128) / 4, 128, 0, stream>>>(
      mem_bf, kvw_bf, CC, 2 * CC, 128, 1, nullptr, k_bf, vt_bf, nullptr, nullptr);

  // Fused attention: scores + mask + bias (raw out) + softmax + P@V -> ctx_bf
  attn_kernel<<<BB * HH * (NN / 16), 128, 0, stream>>>(
      q_bf, k_bf, vt_bf, bias, mask, out_attn, ctx_bf);

  // Output projection: (4096 x 1024) @ (1024 x 1024)^T + b -> out_x (f32)
  gemm_bf16_kernel<<<(256 * 64) / 4, 128, 0, stream>>>(
      ctx_bf, pw_bf, CC, CC, 64, 2, nullptr, nullptr, nullptr, out_x, proj_b);
}
